// AnomalyAttention_31069793419538
// MI455X (gfx1250) — matrix operand order repaired and checked by reading, not compiled, not run on hardware
//
#include <hip/hip_runtime.h>
#include <hip/hip_bf16.h>
#include <math.h>

#define N_TOK 8192
#define DMODEL 512

typedef __attribute__((ext_vector_type(16))) __bf16 bf16x16;
typedef __attribute__((ext_vector_type(8)))  __bf16 bf16x8;
typedef __attribute__((ext_vector_type(8)))  float  floatx8;

// ---------------------------------------------------------------------------
// WMMA helpers (CDNA5 wave32, 16x16x32 bf16 -> f32 accumulate)
// ---------------------------------------------------------------------------
__device__ inline floatx8 wmma_bf16(bf16x16 a, bf16x16 b, floatx8 c) {
  // 8 args: (neg_a, A, neg_b, B, c_mod, C, reuse_a, reuse_b)
  return __builtin_amdgcn_wmma_f32_16x16x32_bf16(false, a, false, b, (short)0, c,
                                                 false, false);
}

// A-matrix 16x32 bf16 fragment, row-major source.
// Lane l holds row (l&15); K chunks [kb..kb+7] and [kb+16..kb+23], kb=(l>>4)*8.
__device__ inline bf16x16 load_fragA_bf16(const __bf16* __restrict__ base, int row,
                                          int ld, int k0) {
  const int lane = threadIdx.x & 31;
  const __bf16* p =
      base + (size_t)(row + (lane & 15)) * ld + k0 + ((lane >> 4) << 3);
  bf16x8 lo = *(const bf16x8*)p;
  bf16x8 hi = *(const bf16x8*)(p + 16);
  bf16x16 r;
#pragma unroll
  for (int i = 0; i < 8; ++i) {
    r[i] = lo[i];
    r[8 + i] = hi[i];
  }
  return r;
}

// Same A fragment but from an fp32 source, converted to bf16 on the fly.
__device__ inline bf16x16 load_fragA_f32(const float* __restrict__ base, int row,
                                         int ld, int k0) {
  const int lane = threadIdx.x & 31;
  const float* p =
      base + (size_t)(row + (lane & 15)) * ld + k0 + ((lane >> 4) << 3);
  bf16x16 r;
#pragma unroll
  for (int i = 0; i < 8; ++i) {
    r[i] = (__bf16)p[i];
    r[8 + i] = (__bf16)p[16 + i];
  }
  return r;
}

// B-matrix 32x16 bf16 fragment, fed from row-major B^T (bt[n][k]).
// Lane l holds column n=(l&15); K chunks [kb..kb+7] and [kb+16..kb+23], kb=k0+(l>>4)*8.
__device__ inline bf16x16 load_fragB_bf16(const __bf16* __restrict__ bt, int col0,
                                          int ld, int k0) {
  const int lane = threadIdx.x & 31;
  const __bf16* p =
      bt + (size_t)(col0 + (lane & 15)) * ld + k0 + ((lane >> 4) << 3);
  bf16x8 lo = *(const bf16x8*)p;
  bf16x8 hi = *(const bf16x8*)(p + 16);
  bf16x16 r;
#pragma unroll
  for (int i = 0; i < 8; ++i) {
    r[i] = lo[i];
    r[8 + i] = hi[i];
  }
  return r;
}

// ---------------------------------------------------------------------------
// fp32 -> bf16 conversion
// ---------------------------------------------------------------------------
__global__ void cvt_f32_bf16_kernel(const float* __restrict__ src,
                                    __bf16* __restrict__ dst, int n) {
  int i = blockIdx.x * blockDim.x + threadIdx.x;
  if (i < n) dst[i] = (__bf16)src[i];
}

// ---------------------------------------------------------------------------
// sigma[i] = max(x[i,:] . Ws[0,:], 0.001)   (one wave per row)
// ---------------------------------------------------------------------------
__global__ void sigma_kernel(const float* __restrict__ x,
                             const float* __restrict__ Ws,
                             float* __restrict__ sigma) {
  const int wave = threadIdx.x >> 5;
  const int lane = threadIdx.x & 31;
  const int row = blockIdx.x * 8 + wave;
  const float* xr = x + (size_t)row * DMODEL;
  float acc = 0.f;
  for (int k = lane; k < DMODEL; k += 32) acc += xr[k] * Ws[k];
#pragma unroll
  for (int off = 16; off > 0; off >>= 1) acc += __shfl_down(acc, off, 32);
  if (lane == 0) sigma[row] = fmaxf(acc, 0.001f);
}

// ---------------------------------------------------------------------------
// Projections: Q = x Wq^T, K = x Wk^T, V^T (stored transposed) from x Wv^T.
// block = 256 threads (8 waves); wave computes a 32x32 tile (2x2 WMMA tiles).
// grid = (M/128, 512/64, 3)
// ---------------------------------------------------------------------------
__global__ __launch_bounds__(256) void proj_kernel(
    const __bf16* __restrict__ xb, const __bf16* __restrict__ Wqb,
    const __bf16* __restrict__ Wkb, const __bf16* __restrict__ Wvb,
    __bf16* __restrict__ Qb, __bf16* __restrict__ Kb,
    __bf16* __restrict__ Vtb) {
  const int wave = threadIdx.x >> 5;
  const int lane = threadIdx.x & 31;
  const int row0 = blockIdx.x * 128 + (wave >> 1) * 32;
  const int col0 = blockIdx.y * 64 + (wave & 1) * 32;
  const __bf16* W = (blockIdx.z == 0) ? Wqb : ((blockIdx.z == 1) ? Wkb : Wvb);

  floatx8 acc00 = {}, acc01 = {}, acc10 = {}, acc11 = {};
  for (int k0 = 0; k0 < DMODEL; k0 += 32) {
    bf16x16 a0 = load_fragA_bf16(xb, row0, DMODEL, k0);
    bf16x16 a1 = load_fragA_bf16(xb, row0 + 16, DMODEL, k0);
    bf16x16 b0 = load_fragB_bf16(W, col0, DMODEL, k0);
    bf16x16 b1 = load_fragB_bf16(W, col0 + 16, DMODEL, k0);
    acc00 = wmma_bf16(a0, b0, acc00);
    acc01 = wmma_bf16(a0, b1, acc01);
    acc10 = wmma_bf16(a1, b0, acc10);
    acc11 = wmma_bf16(a1, b1, acc11);
  }

  const int n = lane & 15;
  const int mh = (lane >> 4) << 3;
  floatx8 acc[2][2] = {{acc00, acc01}, {acc10, acc11}};
  if (blockIdx.z < 2) {
    __bf16* O = (blockIdx.z == 0) ? Qb : Kb;
#pragma unroll
    for (int ti = 0; ti < 2; ++ti)
#pragma unroll
      for (int tj = 0; tj < 2; ++tj)
#pragma unroll
        for (int r = 0; r < 8; ++r)
          O[(size_t)(row0 + ti * 16 + mh + r) * DMODEL + col0 + tj * 16 + n] =
              (__bf16)acc[ti][tj][r];
  } else {
    // store V transposed: Vt[col][row]
#pragma unroll
    for (int ti = 0; ti < 2; ++ti)
#pragma unroll
      for (int tj = 0; tj < 2; ++tj)
#pragma unroll
        for (int r = 0; r < 8; ++r)
          Vtb[(size_t)(col0 + tj * 16 + n) * N_TOK + row0 + ti * 16 + mh + r] =
              (__bf16)acc[ti][tj][r];
  }
}

// ---------------------------------------------------------------------------
// Sraw = (Q K^T) / sqrt(D)   fp32 output, written into the S output region.
// Wave computes 32x64 (2x4 WMMA tiles): 8 WMMAs per 6 fragment loads.
// block = 256 (8 waves as 4x2) -> 128x128 per WG; grid = (64, 64).
// ---------------------------------------------------------------------------
__global__ __launch_bounds__(256) void qk_kernel(const __bf16* __restrict__ Qb,
                                                 const __bf16* __restrict__ Kb,
                                                 float* __restrict__ Sraw) {
  const int wave = threadIdx.x >> 5;
  const int lane = threadIdx.x & 31;
  const int row0 = blockIdx.x * 128 + (wave >> 1) * 32;
  const int col0 = blockIdx.y * 128 + (wave & 1) * 64;

  floatx8 acc[2][4] = {};
  for (int k0 = 0; k0 < DMODEL; k0 += 32) {
    bf16x16 a0 = load_fragA_bf16(Qb, row0, DMODEL, k0);
    bf16x16 a1 = load_fragA_bf16(Qb, row0 + 16, DMODEL, k0);
#pragma unroll
    for (int tj = 0; tj < 4; ++tj) {
      bf16x16 b = load_fragB_bf16(Kb, col0 + tj * 16, DMODEL, k0);
      acc[0][tj] = wmma_bf16(a0, b, acc[0][tj]);
      acc[1][tj] = wmma_bf16(a1, b, acc[1][tj]);
    }
  }

  const float scale = 0.04419417382415922f;  // 1/sqrt(512)
  const int n = lane & 15;
  const int mh = (lane >> 4) << 3;
#pragma unroll
  for (int ti = 0; ti < 2; ++ti)
#pragma unroll
    for (int tj = 0; tj < 4; ++tj)
#pragma unroll
      for (int r = 0; r < 8; ++r)
        Sraw[(size_t)(row0 + ti * 16 + mh + r) * N_TOK + col0 + tj * 16 + n] =
            acc[ti][tj][r] * scale;
}

// ---------------------------------------------------------------------------
// Row-wise standardize (ddof=1) + softmax, in place. One block per row.
// Row is staged in LDS so global memory is touched once each way.
// ---------------------------------------------------------------------------
__device__ inline float block_reduce_add(float v, float* red) {
  const int t = threadIdx.x;
  red[t] = v;
  __syncthreads();
  for (int s = 128; s > 0; s >>= 1) {
    if (t < s) red[t] += red[t + s];
    __syncthreads();
  }
  float r = red[0];
  __syncthreads();
  return r;
}

__device__ inline float block_reduce_max(float v, float* red) {
  const int t = threadIdx.x;
  red[t] = v;
  __syncthreads();
  for (int s = 128; s > 0; s >>= 1) {
    if (t < s) red[t] = fmaxf(red[t], red[t + s]);
    __syncthreads();
  }
  float r = red[0];
  __syncthreads();
  return r;
}

__global__ __launch_bounds__(256) void std_softmax_kernel(float* __restrict__ S) {
  __shared__ float buf[N_TOK];
  __shared__ float red[256];
  const int t = threadIdx.x;
  float* Srow = S + (size_t)blockIdx.x * N_TOK;

  float lsum = 0.f, lsq = 0.f, lmax = -INFINITY;
  for (int j = t; j < N_TOK; j += 256) {
    float v = Srow[j];
    buf[j] = v;
    lsum += v;
    lsq += v * v;
    lmax = fmaxf(lmax, v);
  }
  float sum = block_reduce_add(lsum, red);
  float sumsq = block_reduce_add(lsq, red);
  float gmax = block_reduce_max(lmax, red);

  const float n = (float)N_TOK;
  float mean = sum / n;
  float var = fmaxf((sumsq - n * mean * mean) / (n - 1.f), 0.f);
  float inv = 1.f / (sqrtf(var) + 1e-5f);
  float zmax = (gmax - mean) * inv;

  float lesum = 0.f;
  for (int j = t; j < N_TOK; j += 256) {
    float e = expf((buf[j] - mean) * inv - zmax);
    buf[j] = e;
    lesum += e;
  }
  float esum = block_reduce_add(lesum, red);
  float rinv = 1.f / esum;
  for (int j = t; j < N_TOK; j += 256) Srow[j] = buf[j] * rinv;
}

// ---------------------------------------------------------------------------
// Gaussian prior. The 1/sqrt(2*pi*sigma) factor cancels in the normalization:
// P[i,j] = exp(-0.5*((j-i)/sigma_i)^2) / rowsum. One block per row.
// ---------------------------------------------------------------------------
__global__ __launch_bounds__(256) void prior_kernel(const float* __restrict__ sigma,
                                                    float* __restrict__ P) {
  __shared__ float red[256];
  const int t = threadIdx.x;
  const int row = blockIdx.x;
  const float inv = 1.f / sigma[row];

  float lsum = 0.f;
  for (int j = t; j < N_TOK; j += 256) {
    float d = (float)(j - row) * inv;
    lsum += expf(-0.5f * d * d);
  }
  float sum = block_reduce_add(lsum, red);
  float rs = 1.f / sum;
  float* Prow = P + (size_t)row * N_TOK;
  for (int j = t; j < N_TOK; j += 256) {
    float d = (float)(j - row) * inv;
    Prow[j] = expf(-0.5f * d * d) * rs;
  }
}

// ---------------------------------------------------------------------------
// Z = S @ V : A = S (fp32 -> bf16 on the fly), B fed from row-major V^T.
// One WG owns a full 64-row x 512-col strip of Z so the 268 MB S matrix is
// streamed from HBM exactly once (it does not fit in the 192 MB L2).
// 8 waves as 2x4; wave computes 32x128 = 2x8 WMMA tiles (128 accum VGPRs).
// grid = (8192/64) = 128.
// ---------------------------------------------------------------------------
__global__ __launch_bounds__(256) void z_kernel(const float* __restrict__ S,
                                                const __bf16* __restrict__ Vtb,
                                                float* __restrict__ Z) {
  const int wave = threadIdx.x >> 5;
  const int lane = threadIdx.x & 31;
  const int row0 = blockIdx.x * 64 + (wave >> 2) * 32;
  const int col0 = (wave & 3) * 128;

  floatx8 acc[2][8] = {};
  for (int k0 = 0; k0 < N_TOK; k0 += 32) {
    // prefetch next K-slab of S (speculative; dropped if OOB)
    if (k0 + 32 < N_TOK) {
      const float* pf =
          S + (size_t)(row0 + (lane & 15)) * N_TOK + (k0 + 32) + ((lane >> 4) << 3);
      __builtin_prefetch(pf, 0, 1);
    }
    bf16x16 a0 = load_fragA_f32(S, row0, N_TOK, k0);
    bf16x16 a1 = load_fragA_f32(S, row0 + 16, N_TOK, k0);
#pragma unroll
    for (int tj = 0; tj < 8; ++tj) {
      bf16x16 b = load_fragB_bf16(Vtb, col0 + tj * 16, N_TOK, k0);
      acc[0][tj] = wmma_bf16(a0, b, acc[0][tj]);
      acc[1][tj] = wmma_bf16(a1, b, acc[1][tj]);
    }
  }

  const int n = lane & 15;
  const int mh = (lane >> 4) << 3;
#pragma unroll
  for (int ti = 0; ti < 2; ++ti)
#pragma unroll
    for (int tj = 0; tj < 8; ++tj)
#pragma unroll
      for (int r = 0; r < 8; ++r)
        Z[(size_t)(row0 + ti * 16 + mh + r) * DMODEL + col0 + tj * 16 + n] =
            acc[ti][tj][r];
}

// ---------------------------------------------------------------------------
// Launch
// ---------------------------------------------------------------------------
extern "C" void kernel_launch(void* const* d_in, const int* in_sizes, int n_in,
                              void* d_out, int out_size, void* d_ws, size_t ws_size,
                              hipStream_t stream) {
  const float* x  = (const float*)d_in[0];
  const float* Wq = (const float*)d_in[1];
  const float* Wk = (const float*)d_in[2];
  const float* Wv = (const float*)d_in[3];
  const float* Ws = (const float*)d_in[4];

  float* out = (float*)d_out;
  float* Z = out;                                   // [8192, 512]
  float* P = Z + (size_t)N_TOK * DMODEL;            // [8192, 8192]
  float* S = P + (size_t)N_TOK * N_TOK;             // [8192, 8192]

  char* ws = (char*)d_ws;
  __bf16* xb  = (__bf16*)ws;  ws += (size_t)N_TOK * DMODEL * 2;
  __bf16* Wqb = (__bf16*)ws;  ws += (size_t)DMODEL * DMODEL * 2;
  __bf16* Wkb = (__bf16*)ws;  ws += (size_t)DMODEL * DMODEL * 2;
  __bf16* Wvb = (__bf16*)ws;  ws += (size_t)DMODEL * DMODEL * 2;
  __bf16* Qb  = (__bf16*)ws;  ws += (size_t)N_TOK * DMODEL * 2;
  __bf16* Kb  = (__bf16*)ws;  ws += (size_t)N_TOK * DMODEL * 2;
  __bf16* Vtb = (__bf16*)ws;  ws += (size_t)DMODEL * N_TOK * 2;
  float*  sig = (float*)ws;   ws += (size_t)N_TOK * 4;

  // fp32 -> bf16 conversions
  {
    int n = N_TOK * DMODEL;
    cvt_f32_bf16_kernel<<<(n + 255) / 256, 256, 0, stream>>>(x, xb, n);
    n = DMODEL * DMODEL;
    cvt_f32_bf16_kernel<<<(n + 255) / 256, 256, 0, stream>>>(Wq, Wqb, n);
    cvt_f32_bf16_kernel<<<(n + 255) / 256, 256, 0, stream>>>(Wk, Wkb, n);
    cvt_f32_bf16_kernel<<<(n + 255) / 256, 256, 0, stream>>>(Wv, Wvb, n);
  }

  sigma_kernel<<<N_TOK / 8, 256, 0, stream>>>(x, Ws, sig);

  proj_kernel<<<dim3(N_TOK / 128, DMODEL / 64, 3), 256, 0, stream>>>(
      xb, Wqb, Wkb, Wvb, Qb, Kb, Vtb);

  qk_kernel<<<dim3(N_TOK / 128, N_TOK / 128), 256, 0, stream>>>(Qb, Kb, S);

  std_softmax_kernel<<<N_TOK, 256, 0, stream>>>(S);

  prior_kernel<<<N_TOK, 256, 0, stream>>>(sig, P);

  z_kernel<<<N_TOK / 64, 256, 0, stream>>>(S, Vtb, Z);
}
